// MSMM_41274635715096
// MI455X (gfx1250) — compile-verified, matrix operand order checked by reading, not run on hardware
//
#include <hip/hip_runtime.h>
#include <math.h>

// ---------------------------------------------------------------------------
// MambaIR-style block for MI455X (gfx1250, wave32, WMMA).
// ALL matmul-shaped ops (3x3 convs via im2col, mamba projections, lin1/lin2)
// use v_wmma_f32_16x16x32_f16 (f16 in, f32 accum).
// Selective scan: 12288 parallel (b,d,n) lanes, shfl-xor reduce over n=16.
// ---------------------------------------------------------------------------

typedef __attribute__((ext_vector_type(8)))  _Float16 v8h;
typedef __attribute__((ext_vector_type(16))) _Float16 v16h;
typedef __attribute__((ext_vector_type(8)))  float    v8f;

#define BSZ   4
#define HID   96
#define IMG   128
#define LL1   (IMG * IMG)        // 16384
#define LL2   ((IMG/4)*(IMG/4))  // 1024
#define DI    192                // d_inner
#define DST   16                 // d_state
#define DTR   6                  // dt_rank
#define NPAD  48                 // x_proj N (38) padded to 48

// GEMM epilogue flags
#define GF_OUT_F16 1
#define GF_GELU    2
#define GF_NCHW    4

// ------------------------- conversion kernels ------------------------------
__global__ void k_f32_to_f16(const float* __restrict__ s, _Float16* __restrict__ d, size_t n) {
  size_t i = (size_t)blockIdx.x * blockDim.x + threadIdx.x;
  if (i < n) d[i] = (_Float16)s[i];
}

// pad x_proj_w (38,192) -> (48,192) f16, zero fill
__global__ void k_pad48_f16(const float* __restrict__ s, _Float16* __restrict__ d) {
  int i = blockIdx.x * blockDim.x + threadIdx.x;
  if (i >= NPAD * DI) return;
  int row = i / DI, col = i % DI;
  d[i] = (_Float16)(row < (DTR + 2 * DST) ? s[row * DI + col] : 0.f);
}

// ------------------------- im2col (3x3, pad=1) -----------------------------
// NCHW f32 (BSZ,Cin,H,W) -> col f16 (BSZ*Ho*Wo, Cin*9) row-major
__global__ void k_im2col(const float* __restrict__ in, _Float16* __restrict__ col,
                         int Cin, int H, int W, int stride) {
  int Ho = (H - 1) / stride + 1;
  int Wo = (W - 1) / stride + 1;
  size_t total = (size_t)BSZ * Ho * Wo * Cin * 9;
  size_t i = (size_t)blockIdx.x * blockDim.x + threadIdx.x;
  if (i >= total) return;
  int kk = i % 9;
  int ci = (i / 9) % Cin;
  size_t m = i / ((size_t)9 * Cin);     // output pixel row: b*Ho*Wo + ho*Wo + wo
  int wo = m % Wo;
  int ho = (m / Wo) % Ho;
  int b  = m / ((size_t)Wo * Ho);
  int hh = ho * stride - 1 + kk / 3;
  int ww = wo * stride - 1 + kk % 3;
  float v = 0.f;
  if (hh >= 0 && hh < H && ww >= 0 && ww < W)
    v = in[(((size_t)b * Cin + ci) * H + hh) * W + ww];
  col[i] = (_Float16)v;
}

// ------------------------- layernorm over last dim 96 ----------------------
// one wave per row; optional f16 and/or f32 outputs
__global__ void k_layernorm96(const float* __restrict__ x, const float* __restrict__ g,
                              const float* __restrict__ bt, float* __restrict__ of32,
                              _Float16* __restrict__ of16, int rows) {
  int row  = blockIdx.x * 8 + (threadIdx.x >> 5);
  int lane = threadIdx.x & 31;
  if (row >= rows) return;
  const float* xr = x + (size_t)row * HID;
  float v0 = xr[lane], v1 = xr[lane + 32], v2 = xr[lane + 64];
  float s = v0 + v1 + v2;
  #pragma unroll
  for (int m = 16; m >= 1; m >>= 1) s += __shfl_xor(s, m, 32);
  float mean = s * (1.f / 96.f);
  float d0 = v0 - mean, d1 = v1 - mean, d2 = v2 - mean;
  float q = d0 * d0 + d1 * d1 + d2 * d2;
  #pragma unroll
  for (int m = 16; m >= 1; m >>= 1) q += __shfl_xor(q, m, 32);
  float rstd = rsqrtf(q * (1.f / 96.f) + 1e-5f);
  float o0 = d0 * rstd * g[lane]      + bt[lane];
  float o1 = d1 * rstd * g[lane + 32] + bt[lane + 32];
  float o2 = d2 * rstd * g[lane + 64] + bt[lane + 64];
  size_t base = (size_t)row * HID;
  if (of16) { of16[base + lane] = (_Float16)o0; of16[base + lane + 32] = (_Float16)o1; of16[base + lane + 64] = (_Float16)o2; }
  if (of32) { of32[base + lane] = o0;           of32[base + lane + 32] = o1;           of32[base + lane + 64] = o2; }
}

// ------------------------- WMMA GEMM ---------------------------------------
// C(M,N) = act(A(M,K) @ W(N,K)^T [+ bias(N)]) ; A,W f16 row-major;
// K%32==0, M%16==0, N%16==0. One wave per 16x16 tile.
// flags: GF_OUT_F16 store f16 | GF_GELU exact gelu | GF_NCHW store (b,N,HW)
__global__ void __launch_bounds__(128)
k_gemm_wmma(const _Float16* __restrict__ A, const _Float16* __restrict__ W,
            const float* __restrict__ bias, void* __restrict__ Cout,
            int M, int N, int K, int flags, int HW) {
  const int ntn = N >> 4;
  int tile = blockIdx.x * 4 + threadIdx.y;          // threadIdx.y == wave id (uniform)
  if (tile >= (M >> 4) * ntn) return;               // wave-uniform guard: EXEC stays all-1
  int tm = tile / ntn, tn = tile % ntn;
  int lane = threadIdx.x;
  int hs = lane >> 4;                               // half-wave select (K offset +8)
  int r  = lane & 15;                               // row (A) / col (B) within tile
  const _Float16* pA = A + (size_t)(tm * 16 + r) * K + hs * 8;
  const _Float16* pB = W + (size_t)(tn * 16 + r) * K + hs * 8;
  v8f acc = {};
  for (int k0 = 0; k0 < K; k0 += 32) {
    v8h a0 = *(const v8h*)(pA);
    v8h a1 = *(const v8h*)(pA + 16);
    v8h b0 = *(const v8h*)(pB);
    v8h b1 = *(const v8h*)(pB + 16);
    pA += 32; pB += 32;
    __builtin_prefetch(pA, 0, 0);                   // global_prefetch_b8 next K tile
    __builtin_prefetch(pB, 0, 0);
    v16h av, bv;
    #pragma unroll
    for (int i = 0; i < 8; ++i) { av[i] = a0[i]; av[i + 8] = a1[i]; bv[i] = b0[i]; bv[i + 8] = b1[i]; }
    acc = __builtin_amdgcn_wmma_f32_16x16x32_f16(false, av, false, bv, (short)0, acc, false, false);
  }
  int nCol = tn * 16 + r;
  float bb = bias ? bias[nCol] : 0.f;
  int m0 = tm * 16 + hs * 8;
  #pragma unroll
  for (int i = 0; i < 8; ++i) {
    float v = acc[i] + bb;
    if (flags & GF_GELU) v = 0.5f * v * (1.f + erff(v * 0.70710678118f));
    int m = m0 + i;
    size_t idx;
    if (flags & GF_NCHW) { int b = m / HW, hw = m % HW; idx = ((size_t)b * N + nCol) * HW + hw; }
    else                 idx = (size_t)m * N + nCol;
    if (flags & GF_OUT_F16) ((_Float16*)Cout)[idx] = (_Float16)v;
    else                    ((float*)Cout)[idx]    = v;
  }
}

// ------------------------- mamba pieces ------------------------------------
// causal depthwise conv1d (width 4) + bias + silu; reads xz f16 (rows,384) x-part
__global__ void k_dwconv_silu(const _Float16* __restrict__ xz, const float* __restrict__ w,
                              const float* __restrict__ cb, float* __restrict__ xc,
                              _Float16* __restrict__ xch, int L, size_t total) {
  size_t i = (size_t)blockIdx.x * blockDim.x + threadIdx.x;
  if (i >= total) return;
  int d = i % DI;
  size_t ml = i / DI;
  int l = ml % L;
  size_t rowb = ml - l;                 // b*L
  float acc = cb[d];
  #pragma unroll
  for (int j = 0; j < 4; ++j) {
    int t = l - 3 + j;
    if (t >= 0) acc += w[d * 4 + j] * (float)xz[(rowb + t) * (2 * DI) + d];
  }
  float s = acc / (1.f + __expf(-acc)); // silu
  xc[i] = s;
  xch[i] = (_Float16)s;
}

// delta = softplus(dt @ dt_proj_w^T + dt_proj_b); dt = dbc[:, 0:6], dbc stride NPAD
__global__ void k_dtproj(const float* __restrict__ dbc, const float* __restrict__ dtw,
                         const float* __restrict__ dtb, float* __restrict__ delta, size_t total) {
  size_t i = (size_t)blockIdx.x * blockDim.x + threadIdx.x;
  if (i >= total) return;
  int d = i % DI;
  size_t m = i / DI;
  const float* r = dbc + m * NPAD;
  float acc = dtb[d];
  #pragma unroll
  for (int j = 0; j < DTR; ++j) acc += r[j] * dtw[d * DTR + j];
  delta[i] = (acc > 20.f) ? acc : log1pf(__expf(acc));
}

// selective scan: one thread per (b,d,n); 16-lane shfl reduce for y
__global__ void k_scan(const float* __restrict__ xc, const float* __restrict__ delta,
                       const float* __restrict__ dbc, const float* __restrict__ A_log,
                       float* __restrict__ y, int L) {
  int chain = blockIdx.x * 16 + (threadIdx.x >> 4);   // 0 .. BSZ*DI-1
  int n = threadIdx.x & 15;
  int b = chain / DI, d = chain % DI;
  float A = -__expf(A_log[d * DST + n]);
  float h = 0.f;
  const float* xcb  = xc    + (size_t)b * L * DI + d;
  const float* dlb  = delta + (size_t)b * L * DI + d;
  const float* dbcb = dbc   + (size_t)b * L * NPAD;
  float* yb = y + (size_t)b * L * DI + d;
  for (int t = 0; t < L; ++t) {
    float u  = xcb[(size_t)t * DI];
    float dt = dlb[(size_t)t * DI];
    float Bv = dbcb[(size_t)t * NPAD + DTR + n];
    float Cv = dbcb[(size_t)t * NPAD + DTR + DST + n];
    h = __expf(dt * A) * h + dt * u * Bv;
    float part = h * Cv;
    part += __shfl_xor(part, 1, 16);
    part += __shfl_xor(part, 2, 16);
    part += __shfl_xor(part, 4, 16);
    part += __shfl_xor(part, 8, 16);
    if (n == 0) yb[(size_t)t * DI] = part;
  }
}

// ygate = (y + xc*D) * silu(z);  z = xz[:, 192:384]
__global__ void k_gate(const float* __restrict__ y, const float* __restrict__ xc,
                       const _Float16* __restrict__ xz, const float* __restrict__ Dp,
                       _Float16* __restrict__ out, size_t total) {
  size_t i = (size_t)blockIdx.x * blockDim.x + threadIdx.x;
  if (i >= total) return;
  int d = i % DI;
  size_t m = i / DI;
  float z = (float)xz[m * (2 * DI) + DI + d];
  float g = z / (1.f + __expf(-z));
  out[i] = (_Float16)((y[i] + xc[i] * Dp[d]) * g);
}

// comb = x1 * y1 + y2 (all flat-identical layouts)
__global__ void k_combine(const float* __restrict__ x1, const float* __restrict__ y1,
                          const float* __restrict__ y2, float* __restrict__ o, size_t total) {
  size_t i = (size_t)blockIdx.x * blockDim.x + threadIdx.x;
  if (i < total) o[i] = x1[i] * y1[i] + y2[i];
}

// ------------------------- channel attention tail --------------------------
__global__ void k_pool(const float* __restrict__ y, float* __restrict__ pool) {
  int bc = blockIdx.x;                       // 0 .. BSZ*HID-1
  const float* p = y + (size_t)bc * LL1;
  float s = 0.f;
  for (int i = threadIdx.x; i < LL1; i += 256) s += p[i];
  __shared__ float sm[256];
  sm[threadIdx.x] = s;
  __syncthreads();
  for (int st = 128; st > 0; st >>= 1) {
    if (threadIdx.x < st) sm[threadIdx.x] += sm[threadIdx.x + st];
    __syncthreads();
  }
  if (threadIdx.x == 0) pool[bc] = sm[0] * (1.f / LL1);
}

__global__ void k_ca(const float* __restrict__ pool, const float* __restrict__ w1,
                     const float* __restrict__ b1, const float* __restrict__ w2,
                     const float* __restrict__ b2, float* __restrict__ a) {
  int b = blockIdx.x;
  int c = threadIdx.x;                        // 96 threads
  __shared__ float t[3];
  if (c < 3) {
    float s = b1[c];
    for (int k = 0; k < HID; ++k) s += w1[c * HID + k] * pool[b * HID + k];
    t[c] = fmaxf(s, 0.f);
  }
  __syncthreads();
  float s = b2[c];
  #pragma unroll
  for (int j = 0; j < 3; ++j) s += w2[c * 3 + j] * t[j];
  a[b * HID + c] = 1.f / (1.f + __expf(-s));
}

__global__ void k_final(const float* __restrict__ inp, const float* __restrict__ skip,
                        const float* __restrict__ a, const float* __restrict__ caby,
                        float* __restrict__ out, size_t total) {
  size_t i = (size_t)blockIdx.x * blockDim.x + threadIdx.x;
  if (i >= total) return;
  size_t bc = i / (size_t)LL1;
  int c = bc % HID;
  out[i] = inp[i] * skip[c] + a[bc] * caby[i];
}

// ---------------------------------------------------------------------------
static inline unsigned cdiv(size_t a, unsigned b) { return (unsigned)((a + b - 1) / b); }

static void run_gemm(hipStream_t s, const _Float16* A, const _Float16* W, const float* bias,
                     void* C, int M, int N, int K, int flags, int HW) {
  int tiles = (M >> 4) * (N >> 4);
  dim3 blk(32, 4);
  k_gemm_wmma<<<(tiles + 3) / 4, blk, 0, s>>>(A, W, bias, C, M, N, K, flags, HW);
}

// 3x3 conv (pad=1) as im2col + WMMA gemm; output NCHW f32
static void run_conv3x3(hipStream_t s, const float* in, const _Float16* w_f16,
                        const float* bias, float* out, _Float16* col,
                        int Cin, int Cout, int H, int W, int stride, int gelu) {
  int Ho = (H - 1) / stride + 1;
  int Wo = (W - 1) / stride + 1;
  int HW = Ho * Wo;
  size_t ctotal = (size_t)BSZ * HW * Cin * 9;
  k_im2col<<<cdiv(ctotal, 256), 256, 0, s>>>(in, col, Cin, H, W, stride);
  int flags = GF_NCHW | (gelu ? GF_GELU : 0);
  run_gemm(s, col, w_f16, bias, out, BSZ * HW, Cout, Cin * 9, flags, HW);
}

static void run_mamba(hipStream_t s, int L,
                      const _Float16* w_in, const _Float16* w_x, const _Float16* w_out,
                      const float* conv_w, const float* conv_b, const float* dt_w,
                      const float* dt_b, const float* A_log, const float* Dp,
                      _Float16* tAh, _Float16* tXZ, float* tXC, float* tDelta,
                      float* tDbc, float* tY, float* out) {
  const int rows = BSZ * L;
  const size_t tDI = (size_t)rows * DI;
  // in_proj: (rows,96) @ (384,96)^T -> xz f16 (rows,384)
  run_gemm(s, tAh, w_in, nullptr, tXZ, rows, 2 * DI, HID, GF_OUT_F16, 0);
  // causal depthwise conv + silu -> xc f32 + f16 copy into tAh
  k_dwconv_silu<<<cdiv(tDI, 256), 256, 0, s>>>(tXZ, conv_w, conv_b, tXC, tAh, L, tDI);
  // x_proj: (rows,192) @ (48,192)^T -> dbc f32 (rows,48) (cols>=38 are zero)
  run_gemm(s, tAh, w_x, nullptr, tDbc, rows, NPAD, DI, 0, 0);
  // delta = softplus(dt_proj)
  k_dtproj<<<cdiv(tDI, 256), 256, 0, s>>>(tDbc, dt_w, dt_b, tDelta, tDI);
  // selective scan
  k_scan<<<(BSZ * DI) / 16, 256, 0, s>>>(tXC, tDelta, tDbc, A_log, tY, L);
  // gate -> f16 A for out_proj
  k_gate<<<cdiv(tDI, 256), 256, 0, s>>>(tY, tXC, tXZ, Dp, tAh, tDI);
  // out_proj: (rows,192) @ (96,192)^T -> out f32 (rows,96)
  run_gemm(s, tAh, w_out, nullptr, out, rows, HID, DI, 0, 0);
}

extern "C" void kernel_launch(void* const* d_in, const int* in_sizes, int n_in,
                              void* d_out, int out_size, void* d_ws, size_t ws_size,
                              hipStream_t stream) {
  (void)in_sizes; (void)n_in; (void)out_size; (void)ws_size;
  // ---- input map (setup_inputs dict order, params flattened in insertion order)
  const float* inp = (const float*)d_in[0];
  const float* scw = (const float*)d_in[1];  const float* scb = (const float*)d_in[2];
  const float* lcw = (const float*)d_in[3];  const float* lcb = (const float*)d_in[4];
  const float* ln1w = (const float*)d_in[5]; const float* ln1b = (const float*)d_in[6];
  const float* ln2w = (const float*)d_in[7]; const float* ln2b = (const float*)d_in[8];
  const float* ln3w = (const float*)d_in[9]; const float* ln3b = (const float*)d_in[10];
  const int M1 = 11, M2 = 20; // in_proj_w, conv_w, conv_b, x_proj_w, dt_proj_w, dt_proj_b, A_log, D, out_proj_w
  const float* skip = (const float*)d_in[29];
  const float* l1w = (const float*)d_in[30]; const float* l1b = (const float*)d_in[31];
  const float* l2w = (const float*)d_in[32]; const float* l2b = (const float*)d_in[33];
  const float* cab1w = (const float*)d_in[34]; const float* cab1b = (const float*)d_in[35];
  const float* cab2w = (const float*)d_in[36]; const float* cab2b = (const float*)d_in[37];
  const float* ca1w = (const float*)d_in[38];  const float* ca1b = (const float*)d_in[39];
  const float* ca2w = (const float*)d_in[40];  const float* ca2b = (const float*)d_in[41];
  float* out = (float*)d_out;

  // ---- workspace arena
  char* cur = (char*)d_ws;
  auto arena = [&](size_t bytes) -> char* {
    char* p = cur; cur += (bytes + 255) & ~(size_t)255; return p;
  };
  // f16 weights
  _Float16* w_in1  = (_Float16*)arena((size_t)2 * DI * HID * 2);
  _Float16* w_x1   = (_Float16*)arena((size_t)NPAD * DI * 2);
  _Float16* w_out1 = (_Float16*)arena((size_t)HID * DI * 2);
  _Float16* w_in2  = (_Float16*)arena((size_t)2 * DI * HID * 2);
  _Float16* w_x2   = (_Float16*)arena((size_t)NPAD * DI * 2);
  _Float16* w_out2 = (_Float16*)arena((size_t)HID * DI * 2);
  _Float16* w_l1   = (_Float16*)arena((size_t)LL1 * LL2 * 2);
  _Float16* w_l2   = (_Float16*)arena((size_t)LL1 * LL2 * 2);
  _Float16* w_sc   = (_Float16*)arena((size_t)HID * HID * 9 * 2);
  _Float16* w_lc   = (_Float16*)arena((size_t)HID * HID * 9 * 2);
  _Float16* w_cab1 = (_Float16*)arena((size_t)32 * HID * 9 * 2);
  _Float16* w_cab2 = (_Float16*)arena((size_t)HID * 32 * 9 * 2);
  // activations
  float*    t_conv1 = (float*)arena((size_t)BSZ * LL1 * HID * 4);
  _Float16* t_Ah    = (_Float16*)arena((size_t)BSZ * LL1 * DI * 2);
  _Float16* t_xz    = (_Float16*)arena((size_t)BSZ * LL1 * 2 * DI * 2);
  float*    t_xc    = (float*)arena((size_t)BSZ * LL1 * DI * 4);
  float*    t_delta = (float*)arena((size_t)BSZ * LL1 * DI * 4);
  float*    t_dbc   = (float*)arena((size_t)BSZ * LL1 * NPAD * 4);
  float*    t_y     = (float*)arena((size_t)BSZ * LL1 * DI * 4);
  float*    t_x1    = (float*)arena((size_t)BSZ * LL1 * HID * 4);
  float*    t_conv2 = (float*)arena((size_t)BSZ * LL2 * HID * 4);
  _Float16* t_x2h   = (_Float16*)arena((size_t)BSZ * LL2 * HID * 2);
  float*    t_y1    = (float*)arena((size_t)BSZ * LL1 * HID * 4);
  float*    t_y2    = (float*)arena((size_t)BSZ * LL1 * HID * 4);
  _Float16* t_col   = (_Float16*)arena((size_t)BSZ * LL1 * HID * 9 * 2); // im2col (max 65536 x 864)
  float*    t_pool  = (float*)arena((size_t)BSZ * HID * 4);
  float*    t_a     = (float*)arena((size_t)BSZ * HID * 4);
  // aliases (dead-buffer reuse)
  float* t_comb = t_conv1;          // combine result (conv1 output dead)
  float* t_ln3  = (float*)t_xz;     // ln3 output f32 (xz dead; 25MB fits in 50MB)
  float* t_mid  = t_delta;          // CAB mid NCHW (B,32,128,128)
  float* t_caby = t_y;              // CAB out NCHW (B,96,128,128)
  float* t_x2   = t_conv2;          // mamba2 output overwrites its LN input buffer

  // ---- weight conversion to f16 (deterministic, every call)
  k_f32_to_f16<<<cdiv((size_t)2*DI*HID, 256), 256, 0, stream>>>((const float*)d_in[M1+0], w_in1, (size_t)2*DI*HID);
  k_pad48_f16 <<<cdiv((size_t)NPAD*DI, 256), 256, 0, stream>>>((const float*)d_in[M1+3], w_x1);
  k_f32_to_f16<<<cdiv((size_t)HID*DI, 256), 256, 0, stream>>>((const float*)d_in[M1+8], w_out1, (size_t)HID*DI);
  k_f32_to_f16<<<cdiv((size_t)2*DI*HID, 256), 256, 0, stream>>>((const float*)d_in[M2+0], w_in2, (size_t)2*DI*HID);
  k_pad48_f16 <<<cdiv((size_t)NPAD*DI, 256), 256, 0, stream>>>((const float*)d_in[M2+3], w_x2);
  k_f32_to_f16<<<cdiv((size_t)HID*DI, 256), 256, 0, stream>>>((const float*)d_in[M2+8], w_out2, (size_t)HID*DI);
  k_f32_to_f16<<<cdiv((size_t)LL1*LL2, 256), 256, 0, stream>>>(l1w, w_l1, (size_t)LL1*LL2);
  k_f32_to_f16<<<cdiv((size_t)LL1*LL2, 256), 256, 0, stream>>>(l2w, w_l2, (size_t)LL1*LL2);
  k_f32_to_f16<<<cdiv((size_t)HID*HID*9, 256), 256, 0, stream>>>(scw, w_sc, (size_t)HID*HID*9);
  k_f32_to_f16<<<cdiv((size_t)HID*HID*9, 256), 256, 0, stream>>>(lcw, w_lc, (size_t)HID*HID*9);
  k_f32_to_f16<<<cdiv((size_t)32*HID*9, 256), 256, 0, stream>>>(cab1w, w_cab1, (size_t)32*HID*9);
  k_f32_to_f16<<<cdiv((size_t)HID*32*9, 256), 256, 0, stream>>>(cab2w, w_cab2, (size_t)HID*32*9);

  // ---- branch 1: small conv (s=1, WMMA) -> LN -> mamba1 (L=16384)
  run_conv3x3(stream, inp, w_sc, scb, t_conv1, t_col, HID, HID, IMG, IMG, 1, 0);
  k_layernorm96<<<(BSZ*LL1)/8, 256, 0, stream>>>(t_conv1, ln1w, ln1b, nullptr, t_Ah, BSZ*LL1);
  run_mamba(stream, LL1, w_in1, w_x1, w_out1,
            (const float*)d_in[M1+1], (const float*)d_in[M1+2], (const float*)d_in[M1+4],
            (const float*)d_in[M1+5], (const float*)d_in[M1+6], (const float*)d_in[M1+7],
            t_Ah, t_xz, t_xc, t_delta, t_dbc, t_y, t_x1);

  // ---- branch 2: strided conv (s=4, WMMA) -> LN -> mamba2 (L=1024)
  run_conv3x3(stream, inp, w_lc, lcb, t_conv2, t_col, HID, HID, IMG, IMG, 4, 0);
  k_layernorm96<<<(BSZ*LL2)/8, 256, 0, stream>>>(t_conv2, ln2w, ln2b, nullptr, t_Ah, BSZ*LL2);
  run_mamba(stream, LL2, w_in2, w_x2, w_out2,
            (const float*)d_in[M2+1], (const float*)d_in[M2+2], (const float*)d_in[M2+4],
            (const float*)d_in[M2+5], (const float*)d_in[M2+6], (const float*)d_in[M2+7],
            t_Ah, t_xz, t_xc, t_delta, t_dbc, t_y, t_x2);

  // ---- lin1/lin2: (b,96,1024) @ (16384,1024)^T, flat M=384,K=1024,N=16384
  k_f32_to_f16<<<cdiv((size_t)BSZ*LL2*HID, 256), 256, 0, stream>>>(t_x2, t_x2h, (size_t)BSZ*LL2*HID);
  run_gemm(stream, t_x2h, w_l1, l1b, t_y1, BSZ*HID, LL1, LL2, 0, 0);
  run_gemm(stream, t_x2h, w_l2, l2b, t_y2, BSZ*HID, LL1, LL2, 0, 0);

  // ---- combine (flat elementwise), LN3
  size_t nflat = (size_t)BSZ * LL1 * HID;
  k_combine<<<cdiv(nflat, 256), 256, 0, stream>>>(t_x1, t_y1, t_y2, t_comb, nflat);
  k_layernorm96<<<(BSZ*LL1)/8, 256, 0, stream>>>(t_comb, ln3w, ln3b, t_ln3, nullptr, BSZ*LL1);

  // ---- CAB: conv 96->32 + gelu (fused in WMMA epilogue), conv 32->96
  run_conv3x3(stream, t_ln3, w_cab1, cab1b, t_mid, t_col, HID, 32, IMG, IMG, 1, 1);
  run_conv3x3(stream, t_mid, w_cab2, cab2b, t_caby, t_col, 32, HID, IMG, IMG, 1, 0);

  // ---- channel attention + residual
  k_pool<<<BSZ*HID, 256, 0, stream>>>(t_caby, t_pool);
  k_ca<<<BSZ, HID, 0, stream>>>(t_pool, ca1w, ca1b, ca2w, ca2b, t_a);
  k_final<<<cdiv(nflat, 256), 256, 0, stream>>>(inp, skip, t_a, t_caby, out, nflat);
}
